// SegmentationDecoder_29678224016160
// MI455X (gfx1250) — compile-verified, matrix operand order checked by reading, not run on hardware
//
#include <hip/hip_runtime.h>
#include <hip/hip_bf16.h>
#include <math.h>

// ---------------------------------------------------------------------------
// Model constants (from reference)
// ---------------------------------------------------------------------------
#define BB   8
#define DD   256
#define NQ   100
#define NDL  6
#define NH   8
#define LL   3
#define PP   4
#define FPN  128
#define NC   3
#define DH   32          // D / NH
#define H0   96
#define H1   48
#define H2   24
#define N0   (H0*H0)     // 9216
#define N1   (H1*H1)     // 2304
#define N2   (H2*H2)     // 576

// ---------------------------------------------------------------------------
// WMMA types & helpers (CDNA5 / gfx1250, wave32)
// ---------------------------------------------------------------------------
typedef __bf16          v16bf   __attribute__((ext_vector_type(16)));
typedef float           v8f     __attribute__((ext_vector_type(8)));
typedef unsigned short  ushort8 __attribute__((ext_vector_type(8)));

union Frag {
    v16bf   v;
    ushort8 u8[2];
};

__device__ __forceinline__ unsigned short f2bf(float f) {
    // round-to-nearest-even f32 -> bf16
    unsigned int u = __float_as_uint(f);
    unsigned int r = u + 0x7FFFu + ((u >> 16) & 1u);
    return (unsigned short)(r >> 16);
}

__device__ __forceinline__ float gelu_f(float x) {
    return 0.5f * x * (1.0f + erff(x * 0.70710678118654752440f));
}

// Epilogue for one 16x16 sub-tile held in a v8f accumulator.
// C layout: row-major MxN, or NCHW ((m/HW)*N + n)*HW + m%HW when OUTCHW.
template<int ACT, int OUTCHW>
__device__ __forceinline__ void tile_epilogue(
    v8f acc, int m0w, int nbase, int lr, int hl,
    int M, int N, const float* __restrict__ bias,
    float alpha, int HW, float* __restrict__ C)
{
    int n = nbase + lr;
    if (n >= N) return;
    float bv = bias ? bias[n] : 0.f;
#pragma unroll
    for (int r = 0; r < 8; ++r) {
        int m = m0w + hl * 8 + r;
        if (m < M) {
            float y = acc[r] * alpha + bv;
            if (ACT == 1) y = gelu_f(y);
            long long idx;
            if (OUTCHW) {
                int img = m / HW, pix = m % HW;
                idx = ((long long)img * N + n) * (long long)HW + pix;
            } else {
                idx = (long long)m * N + n;
            }
            C[idx] = y;
        }
    }
}

// ---------------------------------------------------------------------------
// Batched WMMA GEMM:  C = ACT( alpha * A@B + bias )
//   A: MxK  (ATRANS: element (m,k) at A[k*lda+m], else A[m*lda+k])
//   B: KxN  (BTRANS: element (k,n) at B[n*ldb+k], else B[k*ldb+n])
//
// Block = 128 threads (4 waves), block tile 64(M) x 64(N), K step 32.
// A (64x32) and B (32x64) tiles are cooperatively staged into LDS as bf16,
// laid out in CDNA5 fragment K-order:
//   lA[row*32 + kk], lB[col*32 + kk]
// so each lane's fragment = two contiguous 16-byte LDS loads:
//   element e -> kk = ((e>>3)<<4) + (hl<<3) + (e&7)
//   => u8[0] at base + hl*8, u8[1] at base + 16 + hl*8.
// Wave w computes rows [w*16, w*16+16) x all 64 cols: 1 A-fragment reused
// across 4 back-to-back v_wmma_f32_16x16x32_bf16.
// ---------------------------------------------------------------------------
template<int ATRANS, int BTRANS, int ACT, int OUTCHW>
__global__ __launch_bounds__(128)
void k_gemm(const float* __restrict__ A, const float* __restrict__ Bw,
            const float* __restrict__ bias, float* __restrict__ C,
            int M, int N, int K, int lda, int ldb,
            long long sA, long long sB, long long sC,
            float alpha, int HW)
{
    __shared__ __align__(16) unsigned short lA[64 * 32];
    __shared__ __align__(16) unsigned short lB[64 * 32];

    long long bz = blockIdx.z;
    A  += bz * sA;
    Bw += bz * sB;
    C  += bz * sC;

    const int m0 = blockIdx.y * 64;
    const int n0 = blockIdx.x * 64;
    const int tid  = threadIdx.x;
    const int wv   = tid >> 5;
    const int lane = tid & 31;
    const int hl   = lane >> 4;
    const int lr   = lane & 15;

    const v8f z = {0.f, 0.f, 0.f, 0.f, 0.f, 0.f, 0.f, 0.f};
    v8f acc0 = z, acc1 = z, acc2 = z, acc3 = z;

    const int abase = (wv * 16 + lr) * 32;

    for (int k0 = 0; k0 < K; k0 += 32) {
        // ---- stage A tile (64 rows x 32 k) ----
#pragma unroll 4
        for (int i = tid; i < 64 * 32; i += 128) {
            int row = i >> 5, kk = i & 31;
            int gm = m0 + row, gk = k0 + kk;
            float v = 0.f;
            if (gm < M && gk < K)
                v = ATRANS ? A[(long long)gk * lda + gm]
                           : A[(long long)gm * lda + gk];
            lA[i] = f2bf(v);
        }
        // ---- stage B tile (32 k x 64 cols), coalesced along n ----
#pragma unroll 4
        for (int i = tid; i < 64 * 32; i += 128) {
            int kk = i >> 6, nn = i & 63;
            int gk = k0 + kk, gn = n0 + nn;
            float v = 0.f;
            if (gk < K && gn < N)
                v = BTRANS ? Bw[(long long)gn * ldb + gk]
                           : Bw[(long long)gk * ldb + gn];
            lB[nn * 32 + kk] = f2bf(v);
        }
        __syncthreads();

        // ---- fragments from LDS: branch-free 16B loads ----
        Frag fa;
        fa.u8[0] = *(const ushort8*)&lA[abase + hl * 8];
        fa.u8[1] = *(const ushort8*)&lA[abase + 16 + hl * 8];

        Frag fb;
#define GEMM_SUBTILE(J, ACCV)                                                  \
        {                                                                      \
            int bbase = ((J) * 16 + lr) * 32;                                  \
            fb.u8[0] = *(const ushort8*)&lB[bbase + hl * 8];                   \
            fb.u8[1] = *(const ushort8*)&lB[bbase + 16 + hl * 8];              \
            ACCV = __builtin_amdgcn_wmma_f32_16x16x32_bf16(                    \
                false, fa.v, false, fb.v, (short)0, ACCV, false, false);       \
        }
        GEMM_SUBTILE(0, acc0)
        GEMM_SUBTILE(1, acc1)
        GEMM_SUBTILE(2, acc2)
        GEMM_SUBTILE(3, acc3)
#undef GEMM_SUBTILE
        __syncthreads();
    }

    const int m0w = m0 + wv * 16;
    tile_epilogue<ACT, OUTCHW>(acc0, m0w, n0 +  0, lr, hl, M, N, bias, alpha, HW, C);
    tile_epilogue<ACT, OUTCHW>(acc1, m0w, n0 + 16, lr, hl, M, N, bias, alpha, HW, C);
    tile_epilogue<ACT, OUTCHW>(acc2, m0w, n0 + 32, lr, hl, M, N, bias, alpha, HW, C);
    tile_epilogue<ACT, OUTCHW>(acc3, m0w, n0 + 48, lr, hl, M, N, bias, alpha, HW, C);
}

// ---------------------------------------------------------------------------
// Implicit-GEMM 3x3 conv (pad=1, stride 1), NCHW in/out, OIHW weights.
// Same 64x64 LDS-staged WMMA structure; the im2col gather (with zero padding)
// happens once per element during staging, amortized over the 64-wide N tile.
// M = B*H*W, N = Cout, K = Cin*9, k = ci*9 + ky*3 + kx.
// ---------------------------------------------------------------------------
__global__ __launch_bounds__(128)
void k_conv3x3(const float* __restrict__ X, const float* __restrict__ W,
               float* __restrict__ Y, int Bn, int Cc, int H, int Wd)
{
    __shared__ __align__(16) unsigned short lA[64 * 32];
    __shared__ __align__(16) unsigned short lB[64 * 32];

    const int HW = H * Wd;
    const int M = Bn * HW;
    const int N = Cc;
    const int K = Cc * 9;

    const int m0 = blockIdx.y * 64;
    const int n0 = blockIdx.x * 64;
    const int tid  = threadIdx.x;
    const int wv   = tid >> 5;
    const int lane = tid & 31;
    const int hl   = lane >> 4;
    const int lr   = lane & 15;

    const v8f z = {0.f, 0.f, 0.f, 0.f, 0.f, 0.f, 0.f, 0.f};
    v8f acc0 = z, acc1 = z, acc2 = z, acc3 = z;

    const int abase = (wv * 16 + lr) * 32;

    for (int k0 = 0; k0 < K; k0 += 32) {
        // ---- stage A tile: im2col gather with zero padding ----
#pragma unroll 4
        for (int i = tid; i < 64 * 32; i += 128) {
            int row = i >> 5, kk = i & 31;
            int gm = m0 + row, gk = k0 + kk;
            float v = 0.f;
            if (gm < M && gk < K) {
                int ci = gk / 9;
                int r9 = gk - ci * 9;
                int ky = r9 / 3;
                int kx = r9 - ky * 3;
                int ab  = gm / HW;
                int pix = gm - ab * HW;
                int ay  = pix / Wd;
                int ax  = pix - ay * Wd;
                int yy = ay + ky - 1, xx = ax + kx - 1;
                if (yy >= 0 && yy < H && xx >= 0 && xx < Wd)
                    v = X[(((long long)ab * Cc + ci) * H + yy) * Wd + xx];
            }
            lA[i] = f2bf(v);
        }
        // ---- stage B tile: OIHW weights, k-major per output channel ----
#pragma unroll 4
        for (int i = tid; i < 64 * 32; i += 128) {
            int kk = i >> 6, nn = i & 63;
            int gk = k0 + kk, gn = n0 + nn;
            float v = 0.f;
            if (gk < K && gn < N) {
                int ci = gk / 9;
                int r9 = gk - ci * 9;
                int ky = r9 / 3;
                int kx = r9 - ky * 3;
                v = W[(((long long)gn * Cc + ci) * 3 + ky) * 3 + kx];
            }
            lB[nn * 32 + kk] = f2bf(v);
        }
        __syncthreads();

        Frag fa;
        fa.u8[0] = *(const ushort8*)&lA[abase + hl * 8];
        fa.u8[1] = *(const ushort8*)&lA[abase + 16 + hl * 8];

        Frag fb;
#define CONV_SUBTILE(J, ACCV)                                                  \
        {                                                                      \
            int bbase = ((J) * 16 + lr) * 32;                                  \
            fb.u8[0] = *(const ushort8*)&lB[bbase + hl * 8];                   \
            fb.u8[1] = *(const ushort8*)&lB[bbase + 16 + hl * 8];              \
            ACCV = __builtin_amdgcn_wmma_f32_16x16x32_bf16(                    \
                false, fa.v, false, fb.v, (short)0, ACCV, false, false);       \
        }
        CONV_SUBTILE(0, acc0)
        CONV_SUBTILE(1, acc1)
        CONV_SUBTILE(2, acc2)
        CONV_SUBTILE(3, acc3)
#undef CONV_SUBTILE
        __syncthreads();
    }

    const int m0w = m0 + wv * 16;
    tile_epilogue<0, 1>(acc0, m0w, n0 +  0, lr, hl, M, N, nullptr, 1.0f, HW, Y);
    tile_epilogue<0, 1>(acc1, m0w, n0 + 16, lr, hl, M, N, nullptr, 1.0f, HW, Y);
    tile_epilogue<0, 1>(acc2, m0w, n0 + 32, lr, hl, M, N, nullptr, 1.0f, HW, Y);
    tile_epilogue<0, 1>(acc3, m0w, n0 + 48, lr, hl, M, N, nullptr, 1.0f, HW, Y);
}

// ---------------------------------------------------------------------------
// Elementwise / reduction kernels
// ---------------------------------------------------------------------------
__global__ void k_query_init(float* __restrict__ out,
                             const float* __restrict__ qe,
                             const float* __restrict__ qp)
{
    int t = blockIdx.x * blockDim.x + threadIdx.x;
    if (t >= BB * NQ * DD) return;
    int c = t % DD;
    int q = (t / DD) % NQ;
    out[t] = qe[q * DD + c] + qp[q * DD + c];
}

__global__ void k_refpoints(float* __restrict__ ref,
                            const float* __restrict__ qpos,
                            const float* __restrict__ rw,
                            const float* __restrict__ rb)
{
    int t = blockIdx.x * blockDim.x + threadIdx.x;
    if (t >= NQ * LL * 2) return;
    int q = t / (LL * 2);
    int j = t % (LL * 2);
    float acc = rb[j];
    for (int k = 0; k < DD; ++k)
        acc += qpos[q * DD + k] * rw[k * (LL * 2) + j];
    ref[t] = 1.0f / (1.0f + expf(-acc));
}

__global__ void k_softmax12(float* __restrict__ aw)
{
    int t = blockIdx.x * blockDim.x + threadIdx.x;
    if (t >= BB * NQ * NH) return;
    float* p = aw + (size_t)(t >> 3) * (NH * LL * PP) + (size_t)(t & 7) * (LL * PP);
    float mx = p[0];
    for (int i = 1; i < LL * PP; ++i) mx = fmaxf(mx, p[i]);
    float e[LL * PP], s = 0.f;
    for (int i = 0; i < LL * PP; ++i) { e[i] = expf(p[i] - mx); s += e[i]; }
    float inv = 1.0f / s;
    for (int i = 0; i < LL * PP; ++i) p[i] = e[i] * inv;
}

// Deformable multi-scale bilinear sampling + attention-weighted accumulate.
// One thread per (b, q, h, dh) output element.
__global__ void k_deform(const float* __restrict__ ref,
                         const float* __restrict__ off,
                         const float* __restrict__ aw,
                         const float* __restrict__ v0,
                         const float* __restrict__ v1,
                         const float* __restrict__ v2,
                         float* __restrict__ out)
{
    int t = blockIdx.x * blockDim.x + threadIdx.x;
    if (t >= BB * NQ * NH * DH) return;
    int dh = t & 31;
    int h  = (t >> 5) & 7;
    int q  = (t >> 8) % NQ;
    int b  = t / (NQ * NH * DH);

    const int    Hl[3] = {H0, H1, H2};
    const int    Nl[3] = {N0, N1, N2};
    const float* Vp[3] = {v0, v1, v2};

    const float* offp = off + (size_t)(b * NQ + q) * (NH * LL * PP * 2) + h * (LL * PP * 2);
    const float* awp  = aw  + (size_t)(b * NQ + q) * (NH * LL * PP)     + h * (LL * PP);

    float acc = 0.f;
    for (int l = 0; l < LL; ++l) {
        const int    W  = Hl[l];
        const float  Wf = (float)W;
        const int    nl = Nl[l];
        const float* V  = Vp[l];
        float rx = ref[(q * LL + l) * 2 + 0];
        float ry = ref[(q * LL + l) * 2 + 1];
        for (int p = 0; p < PP; ++p) {
            float ox = offp[l * (PP * 2) + p * 2 + 0];
            float oy = offp[l * (PP * 2) + p * 2 + 1];
            float wa = awp[l * PP + p];
            float xs = (rx + ox / Wf) * Wf - 0.5f;
            float ys = (ry + oy / Wf) * Wf - 0.5f;
            float x0f = floorf(xs), y0f = floorf(ys);
            int   x0 = (int)x0f,   y0 = (int)y0f;
            float fx = xs - x0f,   fy = ys - y0f;
            float s = 0.f;
#pragma unroll
            for (int dxy = 0; dxy < 4; ++dxy) {
                int dx = dxy & 1, dy = dxy >> 1;
                int xi = x0 + dx, yi = y0 + dy;
                float w = (dx ? fx : 1.f - fx) * (dy ? fy : 1.f - fy);
                if (xi >= 0 && xi < W && yi >= 0 && yi < W) {
                    size_t idx = ((size_t)(b * nl + yi * W + xi)) * DD + h * DH + dh;
                    s += V[idx] * w;
                }
            }
            acc += wa * s;
        }
    }
    out[(size_t)(b * NQ + q) * DD + h * DH + dh] = acc;
}

// y = LN(x + res) over D=256, block = 256 threads, one row per block.
__global__ __launch_bounds__(256)
void k_add_ln(float* __restrict__ x, const float* __restrict__ res,
              const float* __restrict__ g, const float* __restrict__ bb)
{
    __shared__ float red[DD];
    int row = blockIdx.x;
    int c = threadIdx.x;
    size_t base = (size_t)row * DD;
    float v = x[base + c] + (res ? res[base + c] : 0.f);
    red[c] = v;
    __syncthreads();
    for (int s = DD / 2; s > 0; s >>= 1) {
        if (c < s) red[c] += red[c + s];
        __syncthreads();
    }
    float mean = red[0] * (1.0f / DD);
    __syncthreads();
    float d = v - mean;
    red[c] = d * d;
    __syncthreads();
    for (int s = DD / 2; s > 0; s >>= 1) {
        if (c < s) red[c] += red[c + s];
        __syncthreads();
    }
    float var = red[0] * (1.0f / DD);
    x[base + c] = d * rsqrtf(var + 1e-5f) * g[c] + bb[c];
}

// eval-mode BN (running stats 0/1) + GELU, in place, NCHW.
__global__ void k_bn_gelu(float* __restrict__ x, const float* __restrict__ g,
                          const float* __restrict__ b, int C, int HW, long long total)
{
    long long t = (long long)blockIdx.x * blockDim.x + threadIdx.x;
    if (t >= total) return;
    int c = (int)((t / HW) % C);
    const float inv = 0.99999500003749969f; // 1/sqrt(1+1e-5)
    x[t] = gelu_f(x[t] * (g[c] * inv) + b[c]);
}

// dst(B,C,Ho,Wo) += bilinear_resize(src(B,C,Hi,Wi)), half-pixel, edge clamp.
__global__ void k_upsample_add(float* __restrict__ dst, const float* __restrict__ src,
                               int Bn, int C, int Ho, int Wo, int Hi, int Wi)
{
    long long total = (long long)Bn * C * Ho * Wo;
    long long t = (long long)blockIdx.x * blockDim.x + threadIdx.x;
    if (t >= total) return;
    int xo = (int)(t % Wo);
    int yo = (int)((t / Wo) % Ho);
    long long bc = t / ((long long)Ho * Wo);

    float sx = (xo + 0.5f) * (float)Wi / (float)Wo - 0.5f;
    float sy = (yo + 0.5f) * (float)Hi / (float)Ho - 0.5f;
    float x0f = floorf(sx), y0f = floorf(sy);
    float fx = sx - x0f, fy = sy - y0f;
    int x0 = (int)x0f, y0 = (int)y0f;
    int x0c = min(max(x0, 0), Wi - 1), x1c = min(max(x0 + 1, 0), Wi - 1);
    int y0c = min(max(y0, 0), Hi - 1), y1c = min(max(y0 + 1, 0), Hi - 1);

    const float* s = src + bc * (long long)Hi * Wi;
    float v = (1.f - fx) * (1.f - fy) * s[(long long)y0c * Wi + x0c]
            +        fx  * (1.f - fy) * s[(long long)y0c * Wi + x1c]
            + (1.f - fx) *        fy  * s[(long long)y1c * Wi + x0c]
            +        fx  *        fy  * s[(long long)y1c * Wi + x1c];
    dst[t] += v;
}

// softmax over queries (axis = q) for attn (B, NQ, Npix); one thread per (b,n).
__global__ void k_softmax_q(float* __restrict__ a, int Npix)
{
    int t = blockIdx.x * blockDim.x + threadIdx.x;
    if (t >= BB * Npix) return;
    int b = t / Npix, n = t % Npix;
    float* p = a + (size_t)b * NQ * Npix + n;
    float mx = -1e30f;
    for (int q = 0; q < NQ; ++q) mx = fmaxf(mx, p[(size_t)q * Npix]);
    float s = 0.f;
    for (int q = 0; q < NQ; ++q) s += expf(p[(size_t)q * Npix] - mx);
    float inv = 1.0f / s;
    for (int q = 0; q < NQ; ++q) {
        float e = expf(p[(size_t)q * Npix] - mx);
        p[(size_t)q * Npix] = e * inv;
    }
}

__global__ void k_add(float* __restrict__ c, const float* __restrict__ a,
                      const float* __restrict__ b, long long n)
{
    long long t = (long long)blockIdx.x * blockDim.x + threadIdx.x;
    if (t < n) c[t] = a[t] + b[t];
}

// ---------------------------------------------------------------------------
// Host orchestration
// ---------------------------------------------------------------------------
static inline dim3 gemm_grid(int M, int N, int batch) {
    return dim3((N + 63) / 64, (M + 63) / 64, batch);
}

extern "C" void kernel_launch(void* const* d_in, const int* in_sizes, int n_in,
                              void* d_out, int out_size, void* d_ws, size_t ws_size,
                              hipStream_t stream)
{
    (void)in_sizes; (void)n_in; (void)out_size; (void)ws_size;

    // ---- inputs (setup_inputs flattening order) ----
    const float* feat[3]   = {(const float*)d_in[0], (const float*)d_in[1], (const float*)d_in[2]};
    const float* q_embed   = (const float*)d_in[3];
    const float* q_pos     = (const float*)d_in[4];
    const float* ref_w     = (const float*)d_in[5];
    const float* ref_b     = (const float*)d_in[6];
    const float* v_w       = (const float*)d_in[7];
    const float* v_b       = (const float*)d_in[8];
    const float* off_w     = (const float*)d_in[9];
    const float* off_b     = (const float*)d_in[10];
    const float* aw_w      = (const float*)d_in[11];
    const float* aw_b      = (const float*)d_in[12];
    const float* out_w     = (const float*)d_in[13];
    const float* out_b     = (const float*)d_in[14];
    const float* n1_g      = (const float*)d_in[15];
    const float* n1_b      = (const float*)d_in[16];
    const float* ffn_w1    = (const float*)d_in[17];
    const float* ffn_b1    = (const float*)d_in[18];
    const float* ffn_w2    = (const float*)d_in[19];
    const float* ffn_b2    = (const float*)d_in[20];
    const float* n2_g      = (const float*)d_in[21];
    const float* n2_b      = (const float*)d_in[22];
    const float* dn_g      = (const float*)d_in[23];
    const float* dn_b      = (const float*)d_in[24];
    const float* lat_w     = (const float*)d_in[25];
    const float* lat_g     = (const float*)d_in[26];
    const float* lat_b     = (const float*)d_in[27];
    const float* out0_w    = (const float*)d_in[28];
    const float* out0_g    = (const float*)d_in[29];
    const float* out0_b    = (const float*)d_in[30];
    const float* fc1_w     = (const float*)d_in[31];
    const float* fc1_g     = (const float*)d_in[32];
    const float* fc1_b     = (const float*)d_in[33];
    const float* fc2_w     = (const float*)d_in[34];
    const float* fc2_g     = (const float*)d_in[35];
    const float* fc2_b     = (const float*)d_in[36];
    const float* q2p_w     = (const float*)d_in[37];
    const float* q2p_b     = (const float*)d_in[38];
    const float* seg1_w    = (const float*)d_in[39];
    const float* seg1_g    = (const float*)d_in[40];
    const float* seg1_b    = (const float*)d_in[41];
    const float* seg2_w    = (const float*)d_in[42];
    const float* seg2_b    = (const float*)d_in[43];

    float* out = (float*)d_out;

    // ---- workspace bump allocator ----
    float* wsf = (float*)d_ws;
    size_t off_ws = 0;
    auto alloc = [&](size_t nflt) { float* p = wsf + off_ws; off_ws += nflt; return p; };

    float* refp = alloc(NQ * LL * 2);
    float* qcur = alloc((size_t)BB * NQ * DD);
    float* t0   = alloc((size_t)BB * NQ * DD);
    float* ffnh = alloc((size_t)BB * NQ * 4 * DD);
    float* offb = alloc((size_t)BB * NQ * NH * LL * PP * 2);
    float* awb  = alloc((size_t)BB * NQ * NH * LL * PP);
    float* accb = alloc((size_t)BB * NQ * DD);
    float* v0   = alloc((size_t)BB * N0 * DD);
    float* v1   = alloc((size_t)BB * N1 * DD);
    float* v2   = alloc((size_t)BB * N2 * DD);
    float* lat0 = alloc((size_t)BB * FPN * N0);
    float* lat1 = alloc((size_t)BB * FPN * N1);
    float* lat2 = alloc((size_t)BB * FPN * N2);
    float* cA   = alloc((size_t)BB * FPN * N0);
    float* cB   = alloc((size_t)BB * FPN * N0);
    float* qp   = alloc((size_t)BB * NQ * FPN);
    float* attn = alloc((size_t)BB * NQ * N0);
    float* modb = alloc((size_t)BB * FPN * N0);

    const int NL[3] = {N0, N1, N2};
    float* vlev[3] = {v0, v1, v2};
    const int MQ = BB * NQ; // 800

    // ---- 1) query init + reference points ----
    k_query_init<<<(BB * NQ * DD + 255) / 256, 256, 0, stream>>>(qcur, q_embed, q_pos);
    k_refpoints<<<(NQ * LL * 2 + 63) / 64, 64, 0, stream>>>(refp, q_pos, ref_w, ref_b);

    // ---- 2) decoder layers ----
    for (int i = 0; i < NDL; ++i) {
        const float* vw  = v_w    + (size_t)i * DD * DD;
        const float* vb  = v_b    + (size_t)i * DD;
        const float* ow  = off_w  + (size_t)i * DD * (NH * LL * PP * 2);
        const float* ob  = off_b  + (size_t)i * (NH * LL * PP * 2);
        const float* awW = aw_w   + (size_t)i * DD * (NH * LL * PP);
        const float* awB = aw_b   + (size_t)i * (NH * LL * PP);
        const float* otw = out_w  + (size_t)i * DD * DD;
        const float* otb = out_b  + (size_t)i * DD;
        const float* f1w = ffn_w1 + (size_t)i * DD * 4 * DD;
        const float* f1b = ffn_b1 + (size_t)i * 4 * DD;
        const float* f2w = ffn_w2 + (size_t)i * 4 * DD * DD;
        const float* f2b = ffn_b2 + (size_t)i * DD;

        // value projections per level
        for (int l = 0; l < LL; ++l) {
            int M = BB * NL[l];
            k_gemm<0, 0, 0, 0><<<gemm_grid(M, DD, 1), 128, 0, stream>>>(
                feat[l], vw, vb, vlev[l], M, DD, DD, DD, DD, 0, 0, 0, 1.0f, 0);
        }
        // offsets & attention weights
        k_gemm<0, 0, 0, 0><<<gemm_grid(MQ, NH * LL * PP * 2, 1), 128, 0, stream>>>(
            qcur, ow, ob, offb, MQ, NH * LL * PP * 2, DD, DD, NH * LL * PP * 2, 0, 0, 0, 1.0f, 0);
        k_gemm<0, 0, 0, 0><<<gemm_grid(MQ, NH * LL * PP, 1), 128, 0, stream>>>(
            qcur, awW, awB, awb, MQ, NH * LL * PP, DD, DD, NH * LL * PP, 0, 0, 0, 1.0f, 0);
        k_softmax12<<<(BB * NQ * NH + 255) / 256, 256, 0, stream>>>(awb);

        // deformable sampling -> accb (B,NQ,D)
        k_deform<<<(BB * NQ * NH * DH + 255) / 256, 256, 0, stream>>>(
            refp, offb, awb, v0, v1, v2, accb);

        // output projection + residual LN
        k_gemm<0, 0, 0, 0><<<gemm_grid(MQ, DD, 1), 128, 0, stream>>>(
            accb, otw, otb, t0, MQ, DD, DD, DD, DD, 0, 0, 0, 1.0f, 0);
        k_add_ln<<<MQ, DD, 0, stream>>>(qcur, t0, n1_g + (size_t)i * DD, n1_b + (size_t)i * DD);

        // FFN
        k_gemm<0, 0, 1, 0><<<gemm_grid(MQ, 4 * DD, 1), 128, 0, stream>>>(
            qcur, f1w, f1b, ffnh, MQ, 4 * DD, DD, DD, 4 * DD, 0, 0, 0, 1.0f, 0);
        k_gemm<0, 0, 0, 0><<<gemm_grid(MQ, DD, 1), 128, 0, stream>>>(
            ffnh, f2w, f2b, t0, MQ, DD, 4 * DD, 4 * DD, DD, 0, 0, 0, 1.0f, 0);
        k_add_ln<<<MQ, DD, 0, stream>>>(qcur, t0, n2_g + (size_t)i * DD, n2_b + (size_t)i * DD);
    }
    // final decoder norm (no residual)
    k_add_ln<<<MQ, DD, 0, stream>>>(qcur, nullptr, dn_g, dn_b);

    // ---- 3) FPN laterals: 1x1 conv (BTRANS, NCHW epilogue) + BN + GELU ----
    float* lats[3] = {lat0, lat1, lat2};
    for (int l = 0; l < LL; ++l) {
        int M = BB * NL[l];
        k_gemm<0, 1, 0, 1><<<gemm_grid(M, FPN, 1), 128, 0, stream>>>(
            feat[l], lat_w + (size_t)l * FPN * DD, nullptr, lats[l],
            M, FPN, DD, DD, DD, 0, 0, 0, 1.0f, NL[l]);
        long long tot = (long long)BB * FPN * NL[l];
        k_bn_gelu<<<(int)((tot + 255) / 256), 256, 0, stream>>>(
            lats[l], lat_g + (size_t)l * FPN, lat_b + (size_t)l * FPN, FPN, NL[l], tot);
    }
    // top-down: lat1 += up(lat2); lat0 += up(lat1)
    {
        long long tot1 = (long long)BB * FPN * N1;
        k_upsample_add<<<(int)((tot1 + 255) / 256), 256, 0, stream>>>(
            lat1, lat2, BB, FPN, H1, H1, H2, H2);
        long long tot0 = (long long)BB * FPN * N0;
        k_upsample_add<<<(int)((tot0 + 255) / 256), 256, 0, stream>>>(
            lat0, lat1, BB, FPN, H0, H0, H1, H1);
    }

    // ---- 4) pixel decoder conv stack (implicit-GEMM WMMA) ----
    long long totP = (long long)BB * FPN * N0;
    dim3 cgrid((FPN + 63) / 64, (BB * N0 + 63) / 64, 1);

    k_conv3x3<<<cgrid, 128, 0, stream>>>(lat0, out0_w, cA, BB, FPN, H0, H0);
    k_bn_gelu<<<(int)((totP + 255) / 256), 256, 0, stream>>>(cA, out0_g, out0_b, FPN, N0, totP);
    k_conv3x3<<<cgrid, 128, 0, stream>>>(cA, fc1_w, cB, BB, FPN, H0, H0);
    k_bn_gelu<<<(int)((totP + 255) / 256), 256, 0, stream>>>(cB, fc1_g, fc1_b, FPN, N0, totP);
    k_conv3x3<<<cgrid, 128, 0, stream>>>(cB, fc2_w, cA, BB, FPN, H0, H0);
    k_bn_gelu<<<(int)((totP + 255) / 256), 256, 0, stream>>>(cA, fc2_g, fc2_b, FPN, N0, totP);
    float* pixel = cA;

    // ---- 5) query-pixel modulation ----
    k_gemm<0, 0, 1, 0><<<gemm_grid(MQ, FPN, 1), 128, 0, stream>>>(
        qcur, q2p_w, q2p_b, qp, MQ, FPN, DD, DD, FPN, 0, 0, 0, 1.0f, 0);

    // attn[b] = qp[b] (100x128) @ pixel[b] (128x9216) / sqrt(128)   (batched)
    k_gemm<0, 0, 0, 0><<<gemm_grid(NQ, N0, BB), 128, 0, stream>>>(
        qp, pixel, nullptr, attn, NQ, N0, FPN, FPN, N0,
        (long long)NQ * FPN, (long long)FPN * N0, (long long)NQ * N0,
        0.08838834764831845f, 0);
    k_softmax_q<<<(BB * N0 + 255) / 256, 256, 0, stream>>>(attn, N0);

    // mod[b] = qp[b]^T (128x100) @ attn[b] (100x9216)  -> (B,128,9216) NCHW
    k_gemm<1, 0, 0, 0><<<gemm_grid(FPN, N0, BB), 128, 0, stream>>>(
        qp, attn, nullptr, modb, FPN, N0, NQ, FPN, N0,
        (long long)NQ * FPN, (long long)NQ * N0, (long long)FPN * N0, 1.0f, 0);

    // enhanced = pixel + mod
    k_add<<<(int)((totP + 255) / 256), 256, 0, stream>>>(cB, pixel, modb, totP);

    // ---- 6) segmentation head ----
    k_conv3x3<<<cgrid, 128, 0, stream>>>(cB, seg1_w, lat0, BB, FPN, H0, H0);
    k_bn_gelu<<<(int)((totP + 255) / 256), 256, 0, stream>>>(lat0, seg1_g, seg1_b, FPN, N0, totP);

    // logits: per-batch 1x1 conv as GEMM (ATRANS over NCHW activations,
    // BTRANS over OI weights), NCHW epilogue straight into d_out.
    k_gemm<1, 1, 0, 1><<<gemm_grid(N0, NC, BB), 128, 0, stream>>>(
        lat0, seg2_w, seg2_b, out, N0, NC, FPN, N0, FPN,
        (long long)FPN * N0, 0, (long long)NC * N0, 1.0f, N0);
}
